// PiNet_74904229642683
// MI455X (gfx1250) — matrix operand order repaired and checked by reading, not compiled, not run on hardware
//
#include <hip/hip_runtime.h>

// ---------------------------------------------------------------------------
// PiNet degree-4 polynomial network for MI455X (gfx1250, wave32, WMMA).
//
// Roofline: 174 GFLOP vs ~60 MB of traffic -> compute bound. Use
// v_wmma_f32_16x16x32_f16 (f16 in, f32 accumulate).
//
// Math restructure: z4[i*1728+j, b] = x[b,i] * z3[j, b], and 1728 = 54*32,
// so every K=32 chunk of the degree-4 term is a z3 fragment scaled by a
// per-column scalar. Only [z1;z2;pad;z3] (K = 160+1728) is materialized in
// LDS, already swizzled into the 32x16 f16 B-fragment layout.
//
// Weights are packed once (pack kernel) into f16 A-fragment-linear layout:
// K order = [w1 | w2 | zero-pad to 160 | w3 | w4 (i-major)] = 707 chunks.
// ---------------------------------------------------------------------------

typedef _Float16 v16h __attribute__((ext_vector_type(16)));
typedef float    v8f  __attribute__((ext_vector_type(8)));

#define B_N      8192
#define O_M      512
#define I_DIM    12
#define NUM_MT   32                   // O / 16 m-tiles
#define BCOLS    32                   // batch columns per workgroup
#define NGROUPS  2                    // n-tiles (of 16 cols) per wave
#define KC12     5                    // ceil((12+144)/32) -> K padded to 160
#define KC3      54                   // 1728 / 32
#define KCZ      (KC12 + KC3)         // 59 chunks resident in LDS per n-group
#define KC_TOT   (KCZ + I_DIM * KC3)  // 707 total K-chunks
#define FRAG_B   (32 * 16)            // halfs per fragment (32 lanes x 16 f16)
#define LDS_Z_BYTES (NGROUPS * KCZ * FRAG_B * 2)  // 120832
#define LDS_X_BYTES (BCOLS * I_DIM * 4)           // 1536

// ---------------------------------------------------------------------------
// Pack fp32 weights -> f16 WMMA A-fragments (fragment-linear).
// A 16x32 f16 order used here (contiguous K, matches B): lane<16 holds row M=lane, K = 0..15;
// lane>=16 holds row M=lane-16, K = 16..31. Element e -> kk below.
// ---------------------------------------------------------------------------
__global__ void __launch_bounds__(256)
pinet_pack(const float* __restrict__ w1, const float* __restrict__ w2,
           const float* __restrict__ w3, const float* __restrict__ w4,
           _Float16* __restrict__ apack) {
  const int idx = blockIdx.x * 256 + threadIdx.x;     // one per (chunk, mtile, lane)
  const int l = idx & 31;
  const int t = (idx >> 5) & 31;
  const int c = idx >> 10;
  if (c >= KC_TOT) return;
  const int row = t * 16 + (l & 15);
  const int kb = (l < 16) ? 0 : 16;
  _Float16 v[16];
#pragma unroll
  for (int e = 0; e < 16; ++e) {
    const int kk = kb + e;
    const int kg = c * 32 + kk;
    float x;
    if (kg < 12)          x = w1[row * 12 + kg];
    else if (kg < 156)    x = w2[row * 144 + (kg - 12)];
    else if (kg < 160)    x = 0.0f;                   // zero pad keeps math exact
    else if (kg < 1888)   x = w3[row * 1728 + (kg - 160)];
    else                  x = w4[(size_t)row * 20736 + (kg - 1888)];
    v[e] = (_Float16)x;
  }
  uint4* dst = (uint4*)(apack + (size_t)idx * 16);    // 32 bytes per lane-fragment
  dst[0] = *(const uint4*)(v);
  dst[1] = *(const uint4*)(v + 8);
}

// ---------------------------------------------------------------------------
// Fused GEMM: out[b, o] = w0[o] + sum over 707 K-chunks of A-frag x B-frag.
// Each wave: 4 M-tiles x 2 N-tiles (8 f32 accumulators). 8 waves = full O=512.
// ---------------------------------------------------------------------------
__global__ void __launch_bounds__(256)
pinet_gemm(const float* __restrict__ input, const float* __restrict__ w0,
           const _Float16* __restrict__ apack, float* __restrict__ out) {
  extern __shared__ char smem[];
  _Float16* zs = (_Float16*)smem;                     // [NGROUPS][KCZ][32][16] f16
  float*    xs = (float*)(smem + LDS_Z_BYTES);        // [BCOLS][12] fp32

  const int tid   = threadIdx.x;
  const int lane  = tid & 31;
  const int wave  = tid >> 5;
  const int nbase = blockIdx.x * BCOLS;
  const int lcol  = lane & 15;                        // column within n-group

  // Stage x for this WG's 32 batch rows (contiguous fp32).
  if (tid < BCOLS * I_DIM) xs[tid] = input[(size_t)nbase * I_DIM + tid];
  __syncthreads();

  // Build [z1;z2;pad;z3] B-fragments in LDS.
  // B 32x16 f16 order used here (contiguous K, same as A): lane l holds column N=l%16, K = (l<16 ? e : 16+e).
  for (int item = tid; item < NGROUPS * KCZ * 32; item += 256) {
    const int l  = item & 31;
    const int cz = (item >> 5) % KCZ;
    const int g  = item / (KCZ * 32);
    const float* xc = xs + (g * 16 + (l & 15)) * I_DIM;
    _Float16 v[16];
    if (cz < KC12) {
      const int k0 = cz * 32 + ((l < 16) ? 0 : 16);
#pragma unroll
      for (int e = 0; e < 16; ++e) {
        const int k = k0 + e;
        float z;
        if (k < 12)       z = xc[k];                               // z1
        else if (k < 156) { const int m = k - 12;                  // z2
                            z = xc[m / 12] * xc[m % 12]; }
        else              z = 0.0f;                                // pad
        v[e] = (_Float16)z;
      }
    } else {
      const int j0 = (cz - KC12) * 32 + ((l < 16) ? 0 : 16);
#pragma unroll
      for (int e = 0; e < 16; ++e) {                               // z3
        const int j = j0 + e;
        v[e] = (_Float16)(xc[j / 144] * xc[(j / 12) % 12] * xc[j % 12]);
      }
    }
    uint4* dst = (uint4*)(zs + (size_t)item * 16);
    dst[0] = *(const uint4*)(v);
    dst[1] = *(const uint4*)(v + 8);
  }
  __syncthreads();

  v8f acc[4][NGROUPS] = {};
  const int mt0 = wave * 4;
  const v16h* ap = (const v16h*)apack;
  const v16h* zv = (const v16h*)zs;

  int c = 0;
  // ---- Stage 1: degrees 1..3 straight from LDS fragments ----
#pragma unroll 1
  for (int cz = 0; cz < KCZ; ++cz, ++c) {
    v16h b[NGROUPS];
#pragma unroll
    for (int g = 0; g < NGROUPS; ++g)
      b[g] = zv[(g * KCZ + cz) * 32 + lane];
    if (c + 8 < KC_TOT)
      __builtin_prefetch((const void*)(ap + ((size_t)(c + 8) * NUM_MT + mt0) * 32 + lane), 0, 1);
#pragma unroll
    for (int mt = 0; mt < 4; ++mt) {
      const v16h a = ap[((size_t)c * NUM_MT + (mt0 + mt)) * 32 + lane];
#pragma unroll
      for (int g = 0; g < NGROUPS; ++g)
        acc[mt][g] = __builtin_amdgcn_wmma_f32_16x16x32_f16(
            false, a, false, b[g], (short)0, acc[mt][g], false, false);
    }
  }
  // ---- Stage 2: degree 4, B-fragment = x_i (per column) * z3 fragment ----
#pragma unroll 1
  for (int i = 0; i < I_DIM; ++i) {
    // Per-lane scale x[col, i], re-read from LDS each i (12x total); avoids a
    // runtime-indexed register array (cndmask chains) and saves ~12 VGPRs.
    _Float16 sc[NGROUPS];
#pragma unroll
    for (int g = 0; g < NGROUPS; ++g)
      sc[g] = (_Float16)xs[(g * 16 + lcol) * I_DIM + i];
#pragma unroll 2
    for (int c3 = 0; c3 < KC3; ++c3, ++c) {
      v16h b[NGROUPS];
#pragma unroll
      for (int g = 0; g < NGROUPS; ++g) {
        const v16h z = zv[(g * KCZ + KC12 + c3) * 32 + lane];
        b[g] = z * sc[g];                           // v_pk_mul_f16, lane-scalar
      }
      if (c + 8 < KC_TOT)
        __builtin_prefetch((const void*)(ap + ((size_t)(c + 8) * NUM_MT + mt0) * 32 + lane), 0, 1);
#pragma unroll
      for (int mt = 0; mt < 4; ++mt) {
        const v16h a = ap[((size_t)c * NUM_MT + (mt0 + mt)) * 32 + lane];
#pragma unroll
        for (int g = 0; g < NGROUPS; ++g)
          acc[mt][g] = __builtin_amdgcn_wmma_f32_16x16x32_f16(
              false, a, false, b[g], (short)0, acc[mt][g], false, false);
      }
    }
  }

  // ---- Epilogue: + w0 bias, store out[b, o] (rows contiguous per lane) ----
  const int rbase = (lane < 16) ? 0 : 8;            // C/D: lane>=16 -> M = 8+r
#pragma unroll
  for (int mt = 0; mt < 4; ++mt) {
    const int mrow = (mt0 + mt) * 16 + rbase;
    float w0v[8];
    *(float4*)(w0v)     = *(const float4*)(w0 + mrow);
    *(float4*)(w0v + 4) = *(const float4*)(w0 + mrow + 4);
#pragma unroll
    for (int g = 0; g < NGROUPS; ++g) {
      const int col = nbase + g * 16 + lcol;
      float r[8];
#pragma unroll
      for (int e = 0; e < 8; ++e) r[e] = acc[mt][g][e] + w0v[e];
      float* op = out + (size_t)col * O_M + mrow;
      *(float4*)(op)     = *(const float4*)(r);
      *(float4*)(op + 4) = *(const float4*)(r + 4);
    }
  }
}

// ---------------------------------------------------------------------------
extern "C" void kernel_launch(void* const* d_in, const int* in_sizes, int n_in,
                              void* d_out, int out_size, void* d_ws, size_t ws_size,
                              hipStream_t stream) {
  (void)in_sizes; (void)n_in; (void)out_size; (void)ws_size;
  const float* input = (const float*)d_in[0];
  const float* w0    = (const float*)d_in[1];
  const float* w1    = (const float*)d_in[2];
  const float* w2    = (const float*)d_in[3];
  const float* w3    = (const float*)d_in[4];
  const float* w4    = (const float*)d_in[5];
  _Float16* apack    = (_Float16*)d_ws;             // 707*1024*32 B = 23.2 MB
  float* out         = (float*)d_out;

  const int pack_items = KC_TOT * NUM_MT * 32;      // 723,968 (exact /256)
  pinet_pack<<<(pack_items + 255) / 256, 256, 0, stream>>>(w1, w2, w3, w4, apack);

  const int lds_bytes = LDS_Z_BYTES + LDS_X_BYTES;  // 122,368 B (<320KB WGP LDS)
  pinet_gemm<<<B_N / BCOLS, 256, lds_bytes, stream>>>(input, w0, apack, out);
}
